// LBPModule_46213848105960
// MI455X (gfx1250) — matrix operand order repaired and checked by reading, not compiled, not run on hardware
//
#include <hip/hip_runtime.h>
#include <hip/hip_fp16.h>

typedef __attribute__((ext_vector_type(16))) _Float16 v16h;
typedef __attribute__((ext_vector_type(8)))  _Float16 v8h;
typedef __attribute__((ext_vector_type(8)))  float    v8f;

constexpr int Bb = 8, CIN = 64, HW = 4096, Rr = 4;
constexpr int NIMG = 32;          // B*R
constexpr int NN   = 128;         // B*R*4 pair-planes
constexpr int AH = 484, AHP = 496;
constexpr int FEAT = 768;
constexpr int MQ = NIMG * HW;     // 131072 q rows
constexpr int MK = NIMG * AHP;    // 15872 k rows (padded)

__device__ __forceinline__ float lk(float v) { return v > 0.f ? v : 0.01f * v; }

// ---------------------------------------------------------------------------
// Generic f16 WMMA GEMM:  C[M,N] = A[M,K] * W[N,K]^T (+bias[N]) (leaky?)
// A row-major f16, W row-major f16 (i.e. B^T), out f32 or f16.
// One 16x16 tile per wave, K-loop step 32, v_wmma_f32_16x16x32_f16.
// ---------------------------------------------------------------------------
__global__ void __launch_bounds__(256)
gemm_wmma(const _Float16* __restrict__ A, long long sA,
          const _Float16* __restrict__ W, long long sW,
          float* __restrict__ Cf, _Float16* __restrict__ Ch, long long sC,
          const float* __restrict__ bias, int biasLen,
          int M, int N, int K, int lda, int ldw, int ldc, int leaky)
{
    int batch = blockIdx.y;
    const _Float16* Ab = A + (size_t)batch * sA;
    const _Float16* Wb = W + (size_t)batch * sW;
    int wave = threadIdx.x >> 5;
    int lane = threadIdx.x & 31;
    int tilesM = M >> 4, tilesN = N >> 4;
    int tile = blockIdx.x * 8 + wave;
    if (tile >= tilesM * tilesN) return;           // wave-uniform: EXEC stays full
    int tm = tile % tilesM, tn = tile / tilesM;
    int m0 = tm << 4, n0 = tn << 4;
    int hs  = lane >> 4;                            // half-wave select
    int l15 = lane & 15;
    const _Float16* arow = Ab + (size_t)(m0 + l15) * lda + 8 * hs;
    const _Float16* wrow = Wb + (size_t)(n0 + l15) * ldw + 8 * hs;

    v8f acc = {};
    for (int k0 = 0; k0 < K; k0 += 32) {
        const v8h* pa = (const v8h*)(arow + k0);
        v8h a0 = pa[0];
        v8h a1 = pa[2];                             // +16 halves
        const v8h* pw = (const v8h*)(wrow + k0);
        v8h b0 = pw[0];
        v8h b1 = pw[2];
        __builtin_prefetch(arow + k0 + 64, 0, 0);
        __builtin_prefetch(wrow + k0 + 64, 0, 0);
        v16h av, bv;
#pragma unroll
        for (int i = 0; i < 8; ++i) {
            av[i] = a0[i]; av[i + 8] = a1[i];
            bv[i] = b0[i]; bv[i + 8] = b1[i];
        }
        acc = __builtin_amdgcn_wmma_f32_16x16x32_f16(
            false, av, false, bv, (short)0, acc, false, false);
    }

    int col = n0 + l15;
    float bval = (bias && col < biasLen) ? bias[col] : 0.0f;
#pragma unroll
    for (int r = 0; r < 8; ++r) {
        int row = m0 + r + 8 * hs;
        float v = acc[r] + bval;
        if (leaky) v = v > 0.f ? v : 0.01f * v;
        if (Cf) Cf[(size_t)batch * sC + (size_t)row * ldc + col] = v;
        else    Ch[(size_t)batch * sC + (size_t)row * ldc + col] = (_Float16)v;
    }
}

// ---------------------------------------------------------------------------
// f32 -> f16 pack with zero padding (rows x cols) -> (drows x dcols)
// ---------------------------------------------------------------------------
__global__ void __launch_bounds__(256)
k_pack_f16(const float* __restrict__ src, _Float16* __restrict__ dst,
           int rows, int cols, int drows, int dcols)
{
    long long idx = (long long)blockIdx.x * 256 + threadIdx.x;
    if (idx >= (long long)drows * dcols) return;
    int r = (int)(idx / dcols), c = (int)(idx % dcols);
    dst[idx] = (r < rows && c < cols) ? (_Float16)src[(size_t)r * cols + c]
                                      : (_Float16)0.f;
}

// clip(x,-150,150) then 3x3 box sum / 9 (zero pad)
__global__ void __launch_bounds__(256)
k_clipbox(const float* __restrict__ x, float* __restrict__ xb)
{
    int idx = blockIdx.x * 256 + threadIdx.x;
    if (idx >= Bb * CIN * HW) return;
    int hw = idx & (HW - 1), p = idx >> 12;
    int y = hw >> 6, xq = hw & 63;
    const float* pl = x + (size_t)p * HW;
    float s = 0.f;
    for (int dy = -1; dy <= 1; ++dy) {
        int yy = y + dy; if (yy < 0 || yy >= 64) continue;
        for (int dx = -1; dx <= 1; ++dx) {
            int xx = xq + dx; if (xx < 0 || xx >= 64) continue;
            float v = pl[yy * 64 + xx];
            s += fminf(150.f, fmaxf(-150.f, v));
        }
    }
    xb[idx] = s * (1.f / 9.f);
}

// 64->4 3x3 conv, pad 1 (direct, weights in LDS)
__global__ void __launch_bounds__(256)
k_conv1(const float* __restrict__ xb, const float* __restrict__ w,
        const float* __restrict__ bias, float* __restrict__ y1)
{
    __shared__ float sw[Rr * CIN * 9];
    for (int i = threadIdx.x; i < Rr * CIN * 9; i += 256) sw[i] = w[i];
    __syncthreads();
    int idx = blockIdx.x * 256 + threadIdx.x;        // b*HW + hw
    if (idx >= Bb * HW) return;
    int b = idx >> 12, hw = idx & (HW - 1);
    int y = hw >> 6, xq = hw & 63;
    float acc[Rr] = {bias[0], bias[1], bias[2], bias[3]};
    for (int c = 0; c < CIN; ++c) {
        const float* pl = xb + ((size_t)b * CIN + c) * HW;
        for (int t = 0; t < 9; ++t) {
            int yy = y + t / 3 - 1, xx = xq + t % 3 - 1;
            if (yy < 0 || yy >= 64 || xx < 0 || xx >= 64) continue;
            float v = pl[yy * 64 + xx];
#pragma unroll
            for (int r = 0; r < Rr; ++r) acc[r] += v * sw[(r * CIN + c) * 9 + t];
        }
    }
    for (int r = 0; r < Rr; ++r) y1[((size_t)(b * Rr + r)) * HW + hw] = acc[r];
}

// 8 fixed LBP difference kernels -> (NN=128, 2, HW)
__global__ void __launch_bounds__(256)
k_lbp(const float* __restrict__ y1, float* __restrict__ lbp)
{
    const int dyA[8] = {-1, 1, -1, 1, -1, 1, 0, 0};
    const int dxA[8] = {-1, 1,  0, 0,  1,-1, 1,-1};
    int idx = blockIdx.x * 256 + threadIdx.x;        // img*8*HW
    if (idx >= NIMG * 8 * HW) return;
    int hw = idx & (HW - 1);
    int c = (idx >> 12) & 7;
    int img = idx >> 15;
    int y = hw >> 6, xq = hw & 63;
    const float* pl = y1 + (size_t)img * HW;
    float center = pl[hw];
    int yy = y + dyA[c], xx = xq + dxA[c];
    float nb = (yy < 0 || yy >= 64 || xx < 0 || xx >= 64) ? 0.f : pl[yy * 64 + xx];
    int n = img * 4 + (c >> 1), j = c & 1;
    lbp[((size_t)(n * 2 + j)) * HW + hw] = center - nb;
}

// per (n,j) plane: min & cd=(max-min)/8
__global__ void __launch_bounds__(256)
k_minmax(const float* __restrict__ lbp, float* __restrict__ xmin, float* __restrict__ cd)
{
    __shared__ float smin[256], smax[256];
    int p = blockIdx.x;
    const float* pl = lbp + (size_t)p * HW;
    float mn = 3.4e38f, mx = -3.4e38f;
    for (int i = threadIdx.x; i < HW; i += 256) {
        float v = pl[i]; mn = fminf(mn, v); mx = fmaxf(mx, v);
    }
    smin[threadIdx.x] = mn; smax[threadIdx.x] = mx; __syncthreads();
    for (int s = 128; s > 0; s >>= 1) {
        if (threadIdx.x < s) {
            smin[threadIdx.x] = fminf(smin[threadIdx.x], smin[threadIdx.x + s]);
            smax[threadIdx.x] = fmaxf(smax[threadIdx.x], smax[threadIdx.x + s]);
        }
        __syncthreads();
    }
    if (threadIdx.x == 0) { xmin[p] = smin[0]; cd[p] = (smax[0] - smin[0]) * 0.125f; }
}

// joint 8x8 histogram via LDS float atomics (matches inclusive-bin semantics)
__global__ void __launch_bounds__(256)
k_hist(const float* __restrict__ lbp, const float* __restrict__ xmin,
       const float* __restrict__ cd, float* __restrict__ hist)
{
    __shared__ float h[64];
    int n = blockIdx.x;
    if (threadIdx.x < 64) h[threadIdx.x] = 0.f;
    __syncthreads();
    float mn0 = xmin[n * 2], c0 = cd[n * 2];
    float mn1 = xmin[n * 2 + 1], c1 = cd[n * 2 + 1];
    const float* p0 = lbp + (size_t)(n * 2) * HW;
    const float* p1 = p0 + HW;
    for (int i = threadIdx.x; i < HW; i += 256) {
        float a = p0[i], b = p1[i];
        unsigned m0 = 0, m1 = 0;
        for (int q = 0; q < 8; ++q) {
            float lo0 = mn0 + q * c0, lo1 = mn1 + q * c1;
            if (a >= lo0 && a <= lo0 + c0) m0 |= 1u << q;
            if (b >= lo1 && b <= lo1 + c1) m1 |= 1u << q;
        }
        float prod = a * b;
        for (int q = 0; q < 8; ++q) if ((m0 >> q) & 1)
            for (int p = 0; p < 8; ++p) if ((m1 >> p) & 1)
                atomicAdd(&h[q * 8 + p], prod);
    }
    __syncthreads();
    if (threadIdx.x < 64) hist[n * 64 + threadIdx.x] = h[threadIdx.x];
}

__global__ void __launch_bounds__(256)
k_total(const float* __restrict__ hist, float* __restrict__ total)
{
    __shared__ float s[256];
    float a = 0.f;
    for (int i = threadIdx.x; i < NN * 64; i += 256) a += hist[i];
    s[threadIdx.x] = a; __syncthreads();
    for (int st = 128; st > 0; st >>= 1) {
        if (threadIdx.x < st) s[threadIdx.x] += s[threadIdx.x + st];
        __syncthreads();
    }
    if (threadIdx.x == 0) total[0] = s[0];
}

// feat (32, 768): [level0, level1, hist/total] per qq, 4 pair-planes per image
__global__ void __launch_bounds__(256)
k_feat(const float* __restrict__ hist, const float* __restrict__ xmin,
       const float* __restrict__ cd, const float* __restrict__ total,
       float* __restrict__ feat)
{
    int idx = blockIdx.x * 256 + threadIdx.x;
    if (idx >= NIMG * FEAT) return;
    int img = idx / FEAT, r = idx % FEAT;
    int np = r / 192, rem = r % 192, qq = rem / 3, comp = rem % 3;
    int n = img * 4 + np;
    float v;
    if (comp < 2) {
        float m1 = cd[n * 2 + comp] + xmin[n * 2 + comp];   // measure row 1
        float lv = (comp == 0 ? (float)(qq >> 3) : (float)(qq & 7)) + 1.f;
        v = lv * m1;
    } else {
        v = hist[n * 64 + qq] / total[0];
    }
    feat[idx] = v;
}

// q-MLP layer1 (9->256, leaky) + p9^T f16 staging (16 padded rows per image)
__global__ void __launch_bounds__(256)
k_qlayer1(const float* __restrict__ y1, const float* __restrict__ w1,
          const float* __restrict__ b1, _Float16* __restrict__ Hq,
          _Float16* __restrict__ p9T)
{
    __shared__ float sw[256 * 9];
    for (int i = threadIdx.x; i < 256 * 9; i += 256) sw[i] = w1[i];
    __syncthreads();
    int idx = blockIdx.x * 256 + threadIdx.x;        // n*HW + hw
    if (idx >= MQ) return;
    int n = idx >> 12, hw = idx & (HW - 1);
    int y = hw >> 6, xq = hw & 63;
    const float* pl = y1 + (size_t)n * HW;
    float pv[9];
#pragma unroll
    for (int t = 0; t < 9; ++t) {
        int yy = y + t / 3 - 1, xx = xq + t % 3 - 1;
        pv[t] = (yy < 0 || yy >= 64 || xx < 0 || xx >= 64) ? 0.f : pl[yy * 64 + xx];
    }
    _Float16* pT = p9T + (size_t)n * 16 * HW + hw;
#pragma unroll
    for (int t = 0; t < 9; ++t) pT[(size_t)t * HW] = (_Float16)pv[t];
#pragma unroll
    for (int t = 9; t < 16; ++t) pT[(size_t)t * HW] = (_Float16)0.f;
    _Float16* out = Hq + (size_t)idx * 256;
    for (int j = 0; j < 256; ++j) {
        float s = b1[j];
#pragma unroll
        for (int t = 0; t < 9; ++t) s += pv[t] * sw[j * 9 + t];
        out[j] = (_Float16)lk(s);
    }
}

// k-MLP layer1: ra_b + leaky, then 9->256 leaky; zero pad rows o>=484
__global__ void __launch_bounds__(256)
k_klayer1(const float* __restrict__ raRaw, const float* __restrict__ ra_b,
          const float* __restrict__ w1, const float* __restrict__ b1,
          _Float16* __restrict__ Hk)
{
    __shared__ float sw[256 * 9];
    for (int i = threadIdx.x; i < 256 * 9; i += 256) sw[i] = w1[i];
    __syncthreads();
    int idx = blockIdx.x * 256 + threadIdx.x;        // n*496 + o
    if (idx >= MK) return;
    int o = idx % AHP;
    _Float16* out = Hk + (size_t)idx * 256;
    if (o >= AH) { for (int j = 0; j < 256; ++j) out[j] = (_Float16)0.f; return; }
    float rv[9]; float bo = ra_b[o];
    const float* src = raRaw + (size_t)idx * 16;
#pragma unroll
    for (int t = 0; t < 9; ++t) rv[t] = lk(src[t] + bo);
    for (int j = 0; j < 256; ++j) {
        float s = b1[j];
#pragma unroll
        for (int t = 0; t < 9; ++t) s += rv[t] * sw[j * 9 + t];
        out[j] = (_Float16)lk(s);
    }
}

// fused softmax over o<484 + rank-1 value contraction; one wave32 per row
__global__ void __launch_bounds__(256)
k_attn(const float* __restrict__ logits, const float* __restrict__ vout,
       float* __restrict__ attn)
{
    int wave = threadIdx.x >> 5, lane = threadIdx.x & 31;
    long long row = (long long)blockIdx.x * 8 + wave;
    if (row >= MQ) return;
    int n = (int)(row >> 12);
    const float* lr = logits + (size_t)row * AHP;
    const float* vr = vout + (size_t)n * AHP;
    float mx = -3.4e38f;
    for (int o = lane; o < AH; o += 32) mx = fmaxf(mx, lr[o]);
    for (int m = 16; m > 0; m >>= 1) mx = fmaxf(mx, __shfl_xor(mx, m, 32));
    float num = 0.f, den = 0.f;
    for (int o = lane; o < AH; o += 32) {
        float e = __expf(lr[o] - mx);
        den += e; num += e * vr[o];
    }
    for (int m = 16; m > 0; m >>= 1) {
        den += __shfl_xor(den, m, 32);
        num += __shfl_xor(num, m, 32);
    }
    if (lane == 0) attn[row] = num / den;
}

// per-plane instance norm
__global__ void __launch_bounds__(256)
k_instnorm(const float* __restrict__ attn, float* __restrict__ xin)
{
    __shared__ float ss[256], sq[256];
    int n = blockIdx.x;
    const float* p = attn + (size_t)n * HW;
    float a = 0.f, b = 0.f;
    for (int i = threadIdx.x; i < HW; i += 256) { float v = p[i]; a += v; b += v * v; }
    ss[threadIdx.x] = a; sq[threadIdx.x] = b; __syncthreads();
    for (int s = 128; s > 0; s >>= 1) {
        if (threadIdx.x < s) { ss[threadIdx.x] += ss[threadIdx.x + s]; sq[threadIdx.x] += sq[threadIdx.x + s]; }
        __syncthreads();
    }
    float m = ss[0] * (1.f / HW);
    float var = sq[0] * (1.f / HW) - m * m;
    float inv = rsqrtf(var + 1e-5f);
    for (int i = threadIdx.x; i < HW; i += 256)
        xin[(size_t)n * HW + i] = (p[i] - m) * inv;
}

// 4->256 3x3 conv (block handles 256 pixels of one (b,co)); 32768 blocks
__global__ void __launch_bounds__(256)
k_conv2(const float* __restrict__ xin, const float* __restrict__ w,
        const float* __restrict__ bias, float* __restrict__ y2)
{
    __shared__ float sw[Rr * 9];
    int co = (blockIdx.x >> 4) & 255;
    int b  = blockIdx.x >> 12;
    if (threadIdx.x < Rr * 9) sw[threadIdx.x] = w[co * Rr * 9 + threadIdx.x];
    __syncthreads();
    int hw = (blockIdx.x & 15) * 256 + threadIdx.x;
    int y = hw >> 6, xq = hw & 63;
    float acc = bias[co];
    const float* base = xin + (size_t)b * Rr * HW;
    for (int r = 0; r < Rr; ++r) {
        const float* pl = base + (size_t)r * HW;
#pragma unroll
        for (int t = 0; t < 9; ++t) {
            int yy = y + t / 3 - 1, xx = xq + t % 3 - 1;
            if (yy < 0 || yy >= 64 || xx < 0 || xx >= 64) continue;
            acc += pl[yy * 64 + xx] * sw[r * 9 + t];
        }
    }
    y2[((size_t)(b * 256 + co)) * HW + hw] = acc;
}

// batch-norm stats over NCHW layout (one block per channel)
__global__ void __launch_bounds__(256)
k_bnstat_nchw(const float* __restrict__ y, int C, float* __restrict__ mean,
              float* __restrict__ inv)
{
    __shared__ float ss[256], sq[256];
    int c = blockIdx.x;
    float a = 0.f, b2 = 0.f;
    for (int i = threadIdx.x; i < Bb * HW; i += 256) {
        int b = i >> 12, hw = i & (HW - 1);
        float v = y[((size_t)(b * C + c)) * HW + hw];
        a += v; b2 += v * v;
    }
    ss[threadIdx.x] = a; sq[threadIdx.x] = b2; __syncthreads();
    for (int s = 128; s > 0; s >>= 1) {
        if (threadIdx.x < s) { ss[threadIdx.x] += ss[threadIdx.x + s]; sq[threadIdx.x] += sq[threadIdx.x + s]; }
        __syncthreads();
    }
    if (threadIdx.x == 0) {
        float m = ss[0] / (float)(Bb * HW);
        float var = sq[0] / (float)(Bb * HW) - m * m;
        mean[c] = m; inv[c] = rsqrtf(var + 1e-5f);
    }
}

// bn stats over row-major (rows, C) layout
__global__ void __launch_bounds__(256)
k_bnstat_rows(const float* __restrict__ y, int C, int rows,
              float* __restrict__ mean, float* __restrict__ inv)
{
    __shared__ float ss[256], sq[256];
    int c = blockIdx.x;
    float a = 0.f, b2 = 0.f;
    for (int i = threadIdx.x; i < rows; i += 256) {
        float v = y[(size_t)i * C + c];
        a += v; b2 += v * v;
    }
    ss[threadIdx.x] = a; sq[threadIdx.x] = b2; __syncthreads();
    for (int s = 128; s > 0; s >>= 1) {
        if (threadIdx.x < s) { ss[threadIdx.x] += ss[threadIdx.x + s]; sq[threadIdx.x] += sq[threadIdx.x + s]; }
        __syncthreads();
    }
    if (threadIdx.x == 0) {
        float m = ss[0] / (float)rows;
        float var = sq[0] / (float)rows - m * m;
        mean[c] = m; inv[c] = rsqrtf(var + 1e-5f);
    }
}

// BN1-apply + leaky fused into f16 im2col for conv3 (rows=(b,hw), cols=(c,kh,kw))
__global__ void __launch_bounds__(256)
k_im2col3(const float* __restrict__ y2, const float* __restrict__ mean,
          const float* __restrict__ inv, const float* __restrict__ g,
          const float* __restrict__ bb, _Float16* __restrict__ A3)
{
    long long idx = (long long)blockIdx.x * 256 + threadIdx.x;
    if (idx >= (long long)32768 * 2304) return;
    int k = (int)(idx % 2304);
    long long m = idx / 2304;
    int c = k / 9, t = k % 9;
    int b = (int)(m >> 12), hw = (int)(m & (HW - 1));
    int y = (hw >> 6) + t / 3 - 1, xq = (hw & 63) + t % 3 - 1;
    float v = 0.f;
    if (y >= 0 && y < 64 && xq >= 0 && xq < 64) {
        float raw = y2[((size_t)(b * 256 + c)) * HW + y * 64 + xq];
        v = lk((raw - mean[c]) * inv[c] * g[c] + bb[c]);
    }
    A3[idx] = (_Float16)v;
}

// BN2 apply + leaky, (rows,128) -> NCHW output
__global__ void __launch_bounds__(256)
k_final(const float* __restrict__ Y3, const float* __restrict__ mean,
        const float* __restrict__ inv, const float* __restrict__ g,
        const float* __restrict__ bb, float* __restrict__ out)
{
    long long idx = (long long)blockIdx.x * 256 + threadIdx.x;
    if (idx >= (long long)Bb * 128 * HW) return;
    int hw = (int)(idx & (HW - 1));
    int co = (int)((idx >> 12) & 127);
    int b  = (int)(idx >> 19);
    float v = Y3[((size_t)(b * HW + hw)) * 128 + co];
    out[idx] = lk((v - mean[co]) * inv[co] * g[co] + bb[co]);
}

// ---------------------------------------------------------------------------
extern "C" void kernel_launch(void* const* d_in, const int* in_sizes, int n_in,
                              void* d_out, int out_size, void* d_ws, size_t ws_size,
                              hipStream_t stream)
{
    (void)in_sizes; (void)n_in; (void)out_size; (void)ws_size;
    const float* x      = (const float*)d_in[0];
    const float* rc_w   = (const float*)d_in[1];
    const float* rc_b   = (const float*)d_in[2];
    const float* conv_w = (const float*)d_in[3];
    const float* conv_b = (const float*)d_in[4];
    const float* bn1_g  = (const float*)d_in[5];
    const float* bn1_b  = (const float*)d_in[6];
    const float* cf_w   = (const float*)d_in[7];
    const float* cf_b   = (const float*)d_in[8];
    const float* bn2_g  = (const float*)d_in[9];
    const float* bn2_b  = (const float*)d_in[10];
    const float* q_w1   = (const float*)d_in[11];
    const float* q_b1   = (const float*)d_in[12];
    const float* q_w2   = (const float*)d_in[13];
    const float* q_b2   = (const float*)d_in[14];
    const float* k_w1   = (const float*)d_in[15];
    const float* k_b1   = (const float*)d_in[16];
    const float* k_w2   = (const float*)d_in[17];
    const float* k_b2   = (const float*)d_in[18];
    const float* v_w1   = (const float*)d_in[19];
    const float* v_b1   = (const float*)d_in[20];
    const float* v_w2   = (const float*)d_in[21];
    const float* v_b2   = (const float*)d_in[22];
    const float* ra_w   = (const float*)d_in[23];
    const float* ra_b   = (const float*)d_in[24];
    float* out = (float*)d_out;

    size_t off = 0;
    auto alloc = [&](size_t bytes) -> char* {
        char* p = (char*)d_ws + off;
        off = (off + bytes + 255) & ~(size_t)255;
        return p;
    };
    // overlaid regions (lifetimes disjoint)
    char* SH = alloc(32ull * 4096 * AHP * 4);           // logits f32 / A3 f16
    float*    logits = (float*)SH;
    _Float16* A3     = (_Float16*)SH;
    char* R1 = alloc((size_t)MQ * 256 * 2);             // Hq f16 / y2 f32
    _Float16* Hq = (_Float16*)R1;
    float*    y2 = (float*)R1;
    char* R2 = alloc((size_t)Bb * CIN * HW * 4);        // xb f32 / Hk f16
    float*    xb = (float*)R2;
    _Float16* Hk = (_Float16*)R2;
    char* R3 = alloc((size_t)NN * 2 * HW * 4);          // lbp f32 / p9T f16
    float*    lbp = (float*)R3;
    _Float16* p9T = (_Float16*)R3;

    float* y1     = (float*)alloc((size_t)NIMG * HW * 4);
    float* xminb  = (float*)alloc(256 * 4);
    float* cdb    = (float*)alloc(256 * 4);
    float* hist   = (float*)alloc(NN * 64 * 4);
    float* total  = (float*)alloc(256);
    float* feat   = (float*)alloc(NIMG * FEAT * 4);
    _Float16* featf = (_Float16*)alloc((size_t)NIMG * FEAT * 2);
    _Float16* vw1f = (_Float16*)alloc((size_t)256 * 768 * 2);
    _Float16* vw2f = (_Float16*)alloc((size_t)AHP * 256 * 2);
    _Float16* qw2f = (_Float16*)alloc((size_t)64 * 256 * 2);
    _Float16* kw2f = (_Float16*)alloc((size_t)64 * 256 * 2);
    _Float16* rawf = (_Float16*)alloc((size_t)AHP * 4096 * 2);
    _Float16* cfwf = (_Float16*)alloc((size_t)128 * 2304 * 2);
    _Float16* vhid = (_Float16*)alloc((size_t)32 * 256 * 2);
    float* vout    = (float*)alloc((size_t)32 * AHP * 4);
    float* raRaw   = (float*)alloc((size_t)32 * AHP * 16 * 4);
    _Float16* Qm   = (_Float16*)alloc((size_t)MQ * 64 * 2);
    _Float16* Km   = (_Float16*)alloc((size_t)MK * 64 * 2);
    float* attn    = (float*)alloc((size_t)MQ * 4);
    float* xin     = (float*)alloc((size_t)MQ * 4);
    float* m1 = (float*)alloc(256 * 4);
    float* i1 = (float*)alloc(256 * 4);
    float* Y3 = (float*)alloc((size_t)32768 * 128 * 4);
    float* m2 = (float*)alloc(128 * 4);
    float* i2 = (float*)alloc(128 * 4);

    auto gemm = [&](const _Float16* A, long long sA, const _Float16* W, long long sW,
                    float* Cf, _Float16* Ch, long long sC, const float* bias,
                    int biasLen, int M, int N, int K, int lda, int ldw, int ldc,
                    int leaky, int batch) {
        int tiles = (M / 16) * (N / 16);
        dim3 grid((tiles + 7) / 8, batch);
        gemm_wmma<<<grid, 256, 0, stream>>>(A, sA, W, sW, Cf, Ch, sC,
                                            bias, biasLen, M, N, K, lda, ldw, ldc, leaky);
    };
    auto pack = [&](const float* s, _Float16* d, int r, int c, int dr, int dc) {
        long long tot = (long long)dr * dc;
        k_pack_f16<<<dim3((unsigned)((tot + 255) / 256)), 256, 0, stream>>>(s, d, r, c, dr, dc);
    };

    // weight staging (f16, padded)
    pack(v_w1, vw1f, 256, 768, 256, 768);
    pack(v_w2, vw2f, AH, 256, AHP, 256);
    pack(q_w2, qw2f, 64, 256, 64, 256);
    pack(k_w2, kw2f, 64, 256, 64, 256);
    pack(ra_w, rawf, AH, 4096, AHP, 4096);
    pack(cf_w, cfwf, 128, 2304, 128, 2304);

    // front-end
    k_clipbox<<<dim3(Bb * CIN * HW / 256), 256, 0, stream>>>(x, xb);
    k_conv1<<<dim3(Bb * HW / 256), 256, 0, stream>>>(xb, rc_w, rc_b, y1);
    k_lbp<<<dim3(NIMG * 8 * HW / 256), 256, 0, stream>>>(y1, lbp);
    k_minmax<<<dim3(NN * 2), 256, 0, stream>>>(lbp, xminb, cdb);
    k_hist<<<dim3(NN), 256, 0, stream>>>(lbp, xminb, cdb, hist);
    k_total<<<dim3(1), 256, 0, stream>>>(hist, total);
    k_feat<<<dim3((NIMG * FEAT + 255) / 256), 256, 0, stream>>>(hist, xminb, cdb, total, feat);
    pack(feat, featf, NIMG, FEAT, NIMG, FEAT);

    // v MLP: (32,768)->(32,256)->(32,496)
    gemm(featf, 0, vw1f, 0, nullptr, vhid, 0, v_b1, 256, 32, 256, 768, 768, 768, 256, 1, 1);
    gemm(vhid, 0, vw2f, 0, vout, nullptr, 0, v_b2, AH, 32, AHP, 256, 256, 256, AHP, 1, 1);

    // q layer1 + p9T staging (p9T overlays lbp, dead by now)
    k_qlayer1<<<dim3(MQ / 256), 256, 0, stream>>>(y1, q_w1, q_b1, Hq, p9T);

    // ra = ra_w(496x4096) @ p9[n](4096x16), batched over 32 images
    gemm(rawf, 0, p9T, (long long)16 * 4096, raRaw, nullptr, (long long)AHP * 16,
         nullptr, 0, AHP, 16, 4096, 4096, 4096, 16, 0, 32);

    // k layer1 (Hk overlays xb, dead)
    k_klayer1<<<dim3(MK / 256), 256, 0, stream>>>(raRaw, ra_b, k_w1, k_b1, Hk);

    // layer2 projections to embedding dim 64
    gemm(Hq, 0, qw2f, 0, nullptr, Qm, 0, q_b2, 64, MQ, 64, 256, 256, 256, 64, 1, 1);
    gemm(Hk, 0, kw2f, 0, nullptr, Km, 0, k_b2, 64, MK, 64, 256, 256, 256, 64, 1, 1);

    // attention logits: (4096x64)@(64x496) per image
    gemm(Qm, (long long)HW * 64, Km, (long long)AHP * 64, logits, nullptr,
         (long long)HW * AHP, nullptr, 0, HW, AHP, 64, 64, 64, AHP, 0, 32);

    // fused softmax + value contraction, instance norm
    k_attn<<<dim3(MQ / 8), 256, 0, stream>>>(logits, vout, attn);
    k_instnorm<<<dim3(NIMG), 256, 0, stream>>>(attn, xin);

    // conv2 (y2 overlays Hq, dead) + BN1 stats
    // grid = 32768 blocks: bits [3:0]=pixel-block, [11:4]=co, [14:12]=b
    k_conv2<<<dim3(Bb * 256 * 16), 256, 0, stream>>>(xin, conv_w, conv_b, y2);
    k_bnstat_nchw<<<dim3(256), 256, 0, stream>>>(y2, 256, m1, i1);

    // BN1-apply fused im2col (A3 overlays logits, dead), conv3 GEMM, BN2, output
    {
        long long tot = (long long)32768 * 2304;
        k_im2col3<<<dim3((unsigned)((tot + 255) / 256)), 256, 0, stream>>>(y2, m1, i1, bn1_g, bn1_b, A3);
    }
    gemm(A3, 0, cfwf, 0, Y3, nullptr, 0, cf_b, 128, 32768, 128, 2304, 2304, 2304, 128, 0, 1);
    k_bnstat_rows<<<dim3(128), 256, 0, stream>>>(Y3, 128, 32768, m2, i2);
    k_final<<<dim3((unsigned)(((long long)Bb * 128 * HW + 255) / 256)), 256, 0, stream>>>(
        Y3, m2, i2, bn2_g, bn2_b, out);
}
